// LossWrapper_84696755077147
// MI455X (gfx1250) — matrix operand order repaired and checked by reading, not compiled, not run on hardware
//
#include <hip/hip_runtime.h>
#include <math.h>
#include <stdint.h>

typedef __attribute__((ext_vector_type(16))) _Float16     v16h;
typedef __attribute__((ext_vector_type(8)))  float        v8f;
typedef __attribute__((ext_vector_type(4)))  unsigned int u32x4;
typedef __attribute__((ext_vector_type(8)))  int          i32x8;
typedef __attribute__((ext_vector_type(4)))  int          i32x4;

#define N_CLASS 200000
#define EMB     192
#define BN      192   // B*N = 64*3
#define BB      64
#define NN      3
#define SS      3
#define S_SC    30.0f
// margin constants (MARGIN = 0.2)
#define COS_M 0.9800665778412416f    // cos(0.2)
#define SIN_M 0.19866933079506122f   // sin(0.2)
#define TH_C  (-0.9800665778412416f) // cos(pi - 0.2)
#define MM_C  0.039733866159012246f  // sin(pi - 0.2) * 0.2

#define CLS_PER_BLOCK 64
// LDS f32 slab with 1 DWORD of padding after every 64 DWORDs (matches the
// TDM pad_interval=64dw / pad_amount=1dw setting).  Row stride = 195 f32,
// 195 % 64 = 3, gcd(3,64)=1 -> 16 consecutive class rows hit distinct banks.
#define WSLAB_F32 (CLS_PER_BLOCK * EMB + (CLS_PER_BLOCK * EMB) / 64)  // 12480

__device__ __forceinline__ int pidx(int n) { return n + (n >> 6); }

union Frag  { uint32_t u[8]; v16h h; };
union BFrag { uint4 q[2]; uint32_t u[8]; v16h h; };
union Pack2 { _Float16 h[2]; uint32_t u; };

#if defined(__gfx1250__) && __has_builtin(__builtin_amdgcn_tensor_load_to_lds)
#  define HAVE_TDM 1
#  if __has_include(<hip/amd_detail/amd_gfx1250_TDM.h>)
#    define TDM_6ARG 1
#  endif
#endif

// ---------------------------------------------------------------------------
// Kernel 1: normalize pred_embs, emit WMMA B-fragments in per-lane order so
// the GEMM kernel loads each 32x16 f16 fragment as two coalesced b128 loads.
// Fragment element (ct, kc, lane, j):
//   col = lane & 15, hi = lane >= 16, kp = kc*16 + hi*4 + (j < 4 ? j : j + 4)
//   value = half2( x[ct*16+col][2kp], x[ct*16+col][2kp+1] ) / ||x||
// stored at xfrag[ ((ct*6 + kc)*32 + lane)*8 + j ].
// ---------------------------------------------------------------------------
__global__ void prep_x(const float* __restrict__ pred_embs,
                       uint32_t* __restrict__ xfrag,
                       float* __restrict__ sumexp_g)
{
    int bn = threadIdx.x;
    if (bn >= BN) return;
    const float* row = pred_embs + bn * EMB;
    float s = 0.f;
    for (int k = 0; k < EMB; ++k) { float v = row[k]; s += v * v; }
    float rn = 1.f / fmaxf(sqrtf(s), 1e-6f);

    int ct = bn >> 4, col = bn & 15;
    for (int hi = 0; hi < 2; ++hi) {
        int lane = col + hi * 16;
        for (int kc = 0; kc < 6; ++kc)
            for (int j = 0; j < 8; ++j) {
                int kp = kc * 16 + hi * 4 + (j < 4 ? j : j + 4);
                Pack2 p;
                p.h[0] = (_Float16)(row[2 * kp]     * rn);
                p.h[1] = (_Float16)(row[2 * kp + 1] * rn);
                xfrag[((ct * 6 + kc) * 32 + lane) * 8 + j] = p.u;
            }
    }
    sumexp_g[bn] = 0.f;
}

// ---------------------------------------------------------------------------
// Kernel 2 (hot): TDM-fed weight-streaming WMMA GEMM + online fixed-shift LSE.
// Per block: DMA 64 classes x 192 f32 of weight into padded LDS via
// TENSOR_LOAD_TO_LDS (OOB rows auto-zero via tensor_dim1), compute 1/||w||,
// then 8 waves x (16-class A tile held in registers) x 6 bn column tiles.
// All 6 B-fragments of a column tile are preloaded into distinct registers
// so the 6 WMMAs issue back-to-back under partial loadcnt waits.
// sum_c exp(30*cos - 30) is accumulated into sumexp_g[bn].
// ---------------------------------------------------------------------------
__global__ __launch_bounds__(256) void arc_gemm_lse(
    const float* __restrict__ weight,
    const uint4* __restrict__ xfrag,
    float* __restrict__ sumexp_g)
{
    __shared__ float wslab[WSLAB_F32];
    __shared__ float rscale_s[CLS_PER_BLOCK];
    __shared__ float ssum[BN];

    const int t    = threadIdx.x;
    const int base = blockIdx.x * CLS_PER_BLOCK;

    if (t < BN) ssum[t] = 0.f;

#if defined(HAVE_TDM)
    // --- Tensor Data Mover path: one wave issues the 2-D tile DMA ---------
    if (t < 32) {
        uint64_t ga  = (uint64_t)(uintptr_t)(weight + (size_t)base * EMB);
        uint32_t lds = (uint32_t)(uintptr_t)&wslab[0];
        uint32_t rows = (uint32_t)(N_CLASS - base);   // OOB rows read as 0

        u32x4 g0;
        g0[0] = 1u;                                   // count=1, user desc
        g0[1] = lds;                                  // lds_addr
        g0[2] = (uint32_t)ga;                         // global_addr[31:0]
        g0[3] = (uint32_t)(ga >> 32) | (2u << 30);    // global_addr[56:32] | type=2

        i32x8 g1;
        g1[0] = (int)((2u << 16)      // data_size = 4B
                    | (1u << 20)      // pad_enable
                    | (5u << 22));    // pad_interval = 64 DWORDs (amount=1 DWORD)
        g1[1] = (int)((uint32_t)EMB << 16);                 // tensor_dim0[15:0]
        g1[2] = (int)((rows & 0xFFFFu) << 16);              // tensor_dim1[15:0]
        g1[3] = (int)(((rows >> 16) & 0xFFFFu)              // tensor_dim1[31:16]
                    | ((uint32_t)EMB << 16));               // tile_dim0 = 192
        g1[4] = CLS_PER_BLOCK;                              // tile_dim1 = 64
        g1[5] = EMB;                                        // tensor_dim0_stride
        g1[6] = 0;
        g1[7] = 0;

        i32x4 z4 = {0, 0, 0, 0};
#  if defined(TDM_6ARG)
        i32x8 z8 = {0, 0, 0, 0, 0, 0, 0, 0};
        __builtin_amdgcn_tensor_load_to_lds(g0, g1, z4, z4, z8, 0);
#  else
        __builtin_amdgcn_tensor_load_to_lds(g0, g1, z4, z4, 0);
#  endif
        __builtin_amdgcn_s_wait_tensorcnt(0);
    }
    __syncthreads();
#else
    // --- fallback: cooperative synchronous copy into the padded slab ------
    for (int i = 0; i < 12; ++i) {
        int idx = t + 256 * i;          // 0..3071
        int row = idx / 48;
        int q   = idx % 48;
        int cls = base + row;
        float4 w4 = make_float4(0.f, 0.f, 0.f, 0.f);
        if (cls < N_CLASS)
            w4 = ((const float4*)(weight + (size_t)cls * EMB))[q];
        int n = row * EMB + 4 * q;      // n%64 <= 60 -> no pad inside the 4
        int o = pidx(n);
        wslab[o] = w4.x; wslab[o + 1] = w4.y; wslab[o + 2] = w4.z; wslab[o + 3] = w4.w;
    }
    __syncthreads();
#endif

    // Per-class 1/||w|| from the f16-quantized values (matches the GEMM).
    if (t < CLS_PER_BLOCK) {
        float s = 0.f;
        for (int k = 0; k < EMB; ++k) {
            float v = (float)(_Float16)wslab[pidx(t * EMB + k)];
            s += v * v;
        }
        rscale_s[t] = 1.f / fmaxf(sqrtf(s), 1e-6f);
    }
    __syncthreads();

    const int  wave = t >> 5;
    const int  lane = t & 31;
    const int  col  = lane & 15;
    const bool hi   = lane >= 16;
    const int  tileBase = (wave >> 1) * 16;   // waves pair up per 16-class tile
    const int  ctStart  = (wave & 1) * 6;     // each covers 6 bn column tiles

    // A fragments (16 classes x 32 K per K-chunk), f32 LDS -> f16 registers,
    // resident for the whole block.  16-bit A 16x32 layout:
    // lanes 0-15 -> M=lane, K = {2j,2j+1 | j<4} u {16+2(j-4),... | j>=4};
    // lanes 16-31 -> same M, K base shifted by +8.
    Frag afr[6];
    {
        int arow = tileBase + col;
        #pragma unroll
        for (int kc = 0; kc < 6; ++kc) {
            #pragma unroll
            for (int j = 0; j < 8; ++j) {
                int k = kc * 32 + (j < 4 ? 2 * j : 16 + 2 * (j - 4)) + (hi ? 8 : 0);
                int n = arow * EMB + k;     // n even -> pair never crosses pad
                Pack2 p;
                p.h[0] = (_Float16)wslab[pidx(n)];
                p.h[1] = (_Float16)wslab[pidx(n) + 1];
                afr[kc].u[j] = p.u;
            }
        }
    }

    // Scale + validity for this wave's D rows (row M = r + (hi?8:0)).
    float rsc[8], vm[8];
    #pragma unroll
    for (int r = 0; r < 8; ++r) {
        int cl = tileBase + r + (hi ? 8 : 0);
        rsc[r] = rscale_s[cl];
        vm[r]  = ((base + cl) < N_CLASS) ? 1.f : 0.f;
    }

    for (int ci = 0; ci < 6; ++ci) {
        int ct = ctStart + ci;
        // Preload all 6 B-fragments (12 coalesced b128 loads, one clause)
        // into distinct registers so the WMMA chain overlaps the loads.
        BFrag bf[6];
        int ibase = ct * 384 + lane * 2;        // ((ct*6+kc)*32+lane)*2
        #pragma unroll
        for (int kc = 0; kc < 6; ++kc) {
            bf[kc].q[0] = xfrag[ibase + kc * 64];
            bf[kc].q[1] = xfrag[ibase + kc * 64 + 1];
        }
        v8f acc = {};
        #pragma unroll
        for (int kc = 0; kc < 6; ++kc)
            acc = __builtin_amdgcn_wmma_f32_16x16x32_f16(
                false, afr[kc].h, false, bf[kc].h, (short)0, acc, false, false);

        // Online LSE with fixed shift 30: sum exp(30*cos - 30).
        float part = 0.f;
        #pragma unroll
        for (int r = 0; r < 8; ++r) {
            float c = acc[r] * rsc[r];
            c = fminf(fmaxf(c, -1.f + 1e-7f), 1.f - 1e-7f);
            part += vm[r] * __expf(S_SC * c - S_SC);
        }
        // lanes l and l+16 hold the same bn column: combine, then DS atomic.
        part += __shfl_xor(part, 16, 32);
        if (lane < 16) atomicAdd(&ssum[ct * 16 + col], part);
    }
    __syncthreads();
    if (t < BN) atomicAdd(&sumexp_g[t], ssum[t]);
}

// ---------------------------------------------------------------------------
// Kernel 3: cosines at the 576 gathered (b, s, n) label positions, applying
// the same f16 quantization as the GEMM on both operands.
// ---------------------------------------------------------------------------
__global__ void gt_cos(const float* __restrict__ pred_embs,
                       const float* __restrict__ weight,
                       const int* __restrict__ gt_labels,
                       float* __restrict__ cosl)
{
    int t = threadIdx.x;
    if (t >= BB * SS * NN) return;
    int b = t / (SS * NN), s = (t / NN) % SS, n = t % NN;
    int label = gt_labels[b * SS + s];
    const float* xr = pred_embs + (size_t)(b * NN + n) * EMB;
    const float* w  = weight    + (size_t)label * EMB;

    float xs = 0.f;
    for (int k = 0; k < EMB; ++k) { float v = xr[k]; xs += v * v; }
    float rn = 1.f / fmaxf(sqrtf(xs), 1e-6f);

    float dot = 0.f, wsq = 0.f;
    for (int k = 0; k < EMB; ++k) {
        float xh = (float)(_Float16)(xr[k] * rn);
        float wh = (float)(_Float16)w[k];
        dot += xh * wh;
        wsq += wh * wh;
    }
    float c = dot / fmaxf(sqrtf(wsq), 1e-6f);
    c = fminf(fmaxf(c, -1.f + 1e-7f), 1.f - 1e-7f);
    cosl[t] = c;   // [b][s][n]
}

// ---------------------------------------------------------------------------
// Kernel 4: ArcFace cost matrix, 6-permutation matching, count loss, outputs.
// ---------------------------------------------------------------------------
__global__ void finalize(const float* __restrict__ sumexp_g,
                         const float* __restrict__ cosl,
                         const float* __restrict__ pred_ps,
                         float* __restrict__ out)
{
    __shared__ float lse_s[BN];
    __shared__ float lspk_s[BB];
    __shared__ float lcnt_s[BB];
    int t = threadIdx.x;
    if (t < BN) lse_s[t] = __logf(sumexp_g[t]) + S_SC;
    __syncthreads();

    if (t < BB) {
        int b = t;
        float cost[NN][SS];   // cost[n][s] = ce
        for (int s = 0; s < SS; ++s)
            for (int n = 0; n < NN; ++n) {
                float c  = cosl[(b * SS + s) * NN + n];
                float sn = sqrtf(fmaxf(1.f - c * c, 1e-7f));
                float phi = c * COS_M - sn * SIN_M;
                phi = (c > TH_C) ? phi : (c - MM_C);
                float lse  = lse_s[b * NN + n];
                float lmod = lse + log1pf(__expf(S_SC * phi - lse) -
                                          __expf(S_SC * c   - lse));
                cost[n][s] = lmod - S_SC * phi;
            }
        const int P[6][3] = {{0,1,2},{0,2,1},{1,0,2},{1,2,0},{2,0,1},{2,1,0}};
        float best = 1e30f;
        for (int p = 0; p < 6; ++p) {
            float pc = cost[P[p][0]][0] + cost[P[p][1]][1] + cost[P[p][2]][2];
            best = fminf(best, pc);
        }
        lspk_s[b] = best / (float)SS;

        float lc = 0.f;
        for (int i = 0; i < SS; ++i) {
            float p = fminf(fmaxf(pred_ps[b * (SS + 1) + i], 1e-6f), 1.f - 1e-6f);
            lc += log1pf(-p);
        }
        float p3 = fminf(fmaxf(pred_ps[b * (SS + 1) + SS], 1e-6f), 1.f - 1e-6f);
        lc += __logf(p3);
        lcnt_s[b] = -lc / (float)(SS + 1);
    }
    __syncthreads();
    if (t == 0) {
        float a = 0.f, c2 = 0.f;
        for (int b = 0; b < BB; ++b) { a += lspk_s[b]; c2 += lcnt_s[b]; }
        a  /= (float)BB;
        c2 /= (float)BB;
        out[0] = a + 0.1f * c2;   // L_total
        out[1] = a;               // L_spk_total
        out[2] = c2;              // L_cnt_total
    }
}

// ---------------------------------------------------------------------------
extern "C" void kernel_launch(void* const* d_in, const int* in_sizes, int n_in,
                              void* d_out, int out_size, void* d_ws, size_t ws_size,
                              hipStream_t stream)
{
    const float* pred_embs = (const float*)d_in[0];
    const float* pred_ps   = (const float*)d_in[1];
    const int*   gt_labels = (const int*)d_in[2];
    const float* weight    = (const float*)d_in[3];
    float* out = (float*)d_out;

    uint8_t*  ws     = (uint8_t*)d_ws;
    uint32_t* xfrag  = (uint32_t*)ws;              // 12*6*32*8 u32 = 73728 B
    float*    sumexp = (float*)(ws + 73728);       // 192 f32       =   768 B
    float*    cosl   = (float*)(ws + 73728 + 768); // 576 f32       =  2304 B

    prep_x<<<1, 192, 0, stream>>>(pred_embs, xfrag, sumexp);

    int nblk = (N_CLASS + CLS_PER_BLOCK - 1) / CLS_PER_BLOCK;  // 3125
    arc_gemm_lse<<<nblk, 256, 0, stream>>>(weight, (const uint4*)xfrag, sumexp);

    gt_cos<<<1, 576, 0, stream>>>(pred_embs, weight, gt_labels, cosl);
    finalize<<<1, 192, 0, stream>>>(sumexp, cosl, pred_ps, out);
}
